// ScaledDotProductAttention_61134564491714
// MI455X (gfx1250) — matrix operand order repaired and checked by reading, not compiled, not run on hardware
//
#include <hip/hip_runtime.h>
#include <hip/hip_bf16.h>

typedef __attribute__((ext_vector_type(16))) _Float16 v16h;
typedef __attribute__((ext_vector_type(8)))  float    v8f;

#define S_LEN 2048
#define DKV   64
#define HEADS 8
#define QT    16          // q rows per block
#define NTHR  128         // 4 waves
#define SSTR  2052        // LDS row stride (floats): 2052 % 64 = 4 -> 16 rows on distinct banks; mult of 4 -> b128 aligned

__global__ __launch_bounds__(NTHR)
void alibi_attn_kernel(const float* __restrict__ Q,
                       const float* __restrict__ K,
                       const float* __restrict__ V,
                       float* __restrict__ Ow,   // (B,H,S,S) weights
                       float* __restrict__ Oo)   // (B,H,S,DV) output
{
    __shared__ float sc[QT * SSTR];    // 16 x 2048 score/prob block (padded)
    __shared__ float red[QT * 8];
    __shared__ float rowinv[QT];

    const int tid  = threadIdx.x;
    const int wv   = tid >> 5;
    const int lane = tid & 31;
    const int l16  = lane & 15;
    const int hi   = lane >> 4;          // 0 or 1

    const int qt = blockIdx.x & 127;     // S/16 = 128 q tiles
    const int h  = (blockIdx.x >> 7) & 7;
    const int b  = blockIdx.x >> 10;
    const int qbase = qt << 4;

    const float slope = exp2f(-(float)(h + 1));   // ALiBi slope, H=8 -> 2^-(h+1)
    const float NEGINF = -__builtin_inff();
    const size_t bh = (size_t)(b * HEADS + h);
    const float* Qp = Q + bh * (size_t)S_LEN * DKV;
    const float* Kp = K + bh * (size_t)S_LEN * DKV;
    const float* Vp = V + bh * (size_t)S_LEN * DKV;

    // ---- Stage 1: scores = QK^T * scale + alibi, causal mask, into LDS ----
    // A fragment (Q): lane holds row M = l16; element e -> d = e + 8*(e>=8) + 8*hi
    v16h aq0, aq1;
    {
        const float* qrow = Qp + (size_t)(qbase + l16) * DKV;
        #pragma unroll
        for (int e = 0; e < 16; ++e) {
            const int d = e + ((e >= 8) ? 8 : 0) + (hi ? 8 : 0);
            aq0[e] = (_Float16)qrow[d];
            aq1[e] = (_Float16)qrow[d + 32];
        }
    }

    for (int kt = wv; kt <= qt; kt += 4) {          // only causally active k tiles
        const int kbase = kt << 4;
        const float* krow = Kp + (size_t)(kbase + l16) * DKV;
        if (kt + 4 <= qt)
            __builtin_prefetch(Kp + (size_t)(kbase + 64 + l16) * DKV, 0, 0);

        // B fragment (K^T): lane holds col N = kpos = kbase + l16; element e -> d = e + 8*(e>=8) + 8*hi
        v16h bk0, bk1;
        #pragma unroll
        for (int e = 0; e < 16; ++e) {
            const int d = e + ((e >= 8) ? 8 : 0) + (hi ? 8 : 0);
            bk0[e] = (_Float16)krow[d];
            bk1[e] = (_Float16)krow[d + 32];
        }

        v8f c = {};
        c = __builtin_amdgcn_wmma_f32_16x16x32_f16(false, aq0, false, bk0, (short)0, c, false, false);
        c = __builtin_amdgcn_wmma_f32_16x16x32_f16(false, aq1, false, bk1, (short)0, c, false, false);

        const int kcol = kbase + l16;
        #pragma unroll
        for (int r = 0; r < 8; ++r) {
            const int qr = qbase + r + (hi ? 8 : 0);
            float s = c[r] * 0.125f + slope * (float)(kcol - qr);
            if (kcol > qr) s = NEGINF;
            sc[(r + (hi ? 8 : 0)) * SSTR + kcol] = s;
        }
    }
    __syncthreads();

    // ---- Stage 2: rowwise softmax over valid cols [0, q], float4 chunks ----
    const int srow = tid >> 3;           // 0..15
    const int sub  = tid & 7;            // 8 lanes per row, 4-wide chunks
    const int lim  = qbase + srow;       // inclusive causal limit

    float m = NEGINF;
    for (int c4 = sub * 4; c4 <= lim; c4 += 32) {
        const float4 x = *(const float4*)&sc[srow * SSTR + c4];
        m = fmaxf(m, x.x);                                  // c4 <= lim always
        m = fmaxf(m, (c4 + 1 <= lim) ? x.y : NEGINF);
        m = fmaxf(m, (c4 + 2 <= lim) ? x.z : NEGINF);
        m = fmaxf(m, (c4 + 3 <= lim) ? x.w : NEGINF);
    }
    red[srow * 8 + sub] = m;
    __syncthreads();
    #pragma unroll
    for (int i = 0; i < 8; ++i) m = fmaxf(m, red[srow * 8 + i]);
    __syncthreads();

    float psum = 0.f;
    for (int c4 = sub * 4; c4 <= lim; c4 += 32) {
        const float4 x = *(const float4*)&sc[srow * SSTR + c4];
        float4 p;
        p.x = expf(x.x - m);
        p.y = (c4 + 1 <= lim) ? expf(x.y - m) : 0.f;
        p.z = (c4 + 2 <= lim) ? expf(x.z - m) : 0.f;
        p.w = (c4 + 3 <= lim) ? expf(x.w - m) : 0.f;
        psum += (p.x + p.y) + (p.z + p.w);
        *(float4*)&sc[srow * SSTR + c4] = p;
    }
    red[srow * 8 + sub] = psum;
    __syncthreads();
    float tot = 0.f;
    #pragma unroll
    for (int i = 0; i < 8; ++i) tot += red[srow * 8 + i];
    if (sub == 0) rowinv[srow] = 1.0f / tot;
    __syncthreads();

    // ---- Stage 3: normalize P in LDS + stream weights (b128, incl. zero triangle) ----
    const int nkt  = qt + 1;
    const int ktop = ((nkt + 1) & ~1) << 4;   // P cols zero-padded to multiple of 32
    float* wbase = Ow + (bh * (size_t)S_LEN + qbase) * (size_t)S_LEN;
    for (int r = 0; r < QT; ++r) {
        const int rlim = qbase + r;
        const float rv = rowinv[r];
        float* wrow = wbase + (size_t)r * S_LEN;
        for (int c4 = tid * 4; c4 < S_LEN; c4 += NTHR * 4) {
            const float4 x = *(const float4*)&sc[r * SSTR + c4];
            float4 v;
            v.x = (c4 + 0 <= rlim) ? x.x * rv : 0.f;
            v.y = (c4 + 1 <= rlim) ? x.y * rv : 0.f;
            v.z = (c4 + 2 <= rlim) ? x.z * rv : 0.f;
            v.w = (c4 + 3 <= rlim) ? x.w * rv : 0.f;
            if (c4 < ktop)                        // keep P normalized & zero-padded for WMMA
                *(float4*)&sc[r * SSTR + c4] = v;
            *(float4*)&wrow[c4] = v;              // global_store_b128, coalesced
        }
    }
    __syncthreads();

    // ---- Stage 4: O = P x V, each wave owns one 16-wide DV slice ----
    v8f acc = {};
    const int steps = (qt + 2) >> 1;           // ceil((qt+1)/2) K=32 steps
    const int dv = (wv << 4) + l16;
    for (int s2 = 0; s2 < steps; ++s2) {
        const int kb = s2 << 5;
        v16h pa, vb;
        #pragma unroll
        for (int e = 0; e < 16; ++e) {          // A from LDS P: M = l16
            const int kk = kb + e + ((e >= 8) ? 8 : 0) + (hi ? 8 : 0);
            pa[e] = (_Float16)sc[l16 * SSTR + kk];
        }
        #pragma unroll
        for (int e = 0; e < 16; ++e) {          // B from V: N = dv, K = e + 8*(e>=8) + 8*hi
            const int kk = kb + e + ((e >= 8) ? 8 : 0) + (hi ? 8 : 0);
            vb[e] = (_Float16)Vp[(size_t)kk * DKV + dv];
        }
        acc = __builtin_amdgcn_wmma_f32_16x16x32_f16(false, pa, false, vb, (short)0, acc, false, false);
    }

    float* obase = Oo + (bh * (size_t)S_LEN + qbase) * (size_t)DKV;
    #pragma unroll
    for (int r = 0; r < 8; ++r) {
        const int row = r + (hi ? 8 : 0);
        obase[(size_t)row * DKV + dv] = acc[r];
    }
}

extern "C" void kernel_launch(void* const* d_in, const int* in_sizes, int n_in,
                              void* d_out, int out_size, void* d_ws, size_t ws_size,
                              hipStream_t stream) {
    const float* Q = (const float*)d_in[0];
    const float* K = (const float*)d_in[1];
    const float* V = (const float*)d_in[2];
    // d_in[3] = mask (causal, recomputed), d_in[4] = alibi bias (recomputed bitwise)
    float* Oo = (float*)d_out;                                  // (B,H,S,DV) first
    float* Ow = Oo + (size_t)2 * HEADS * S_LEN * DKV;           // then (B,H,S,S)

    const int nblocks = 2 * HEADS * (S_LEN / QT);               // 2048
    alibi_attn_kernel<<<dim3(nblocks), dim3(NTHR), 0, stream>>>(Q, K, V, Ow, Oo);
}
